// MultiHeadAttention_17128329576497
// MI455X (gfx1250) — compile-verified, matrix operand order checked by reading, not run on hardware
//
#include <hip/hip_runtime.h>

// ---------------------------------------------------------------------------
// MI455X (gfx1250) multi-head attention, bf16 WMMA everywhere.
// B=4, T=2048, C=1024, H=16, D=64.
// ---------------------------------------------------------------------------

typedef __attribute__((ext_vector_type(16))) __bf16 v16bf;
typedef __attribute__((ext_vector_type(8)))  float  v8f;

static constexpr int Bsz = 4;
static constexpr int Tln = 2048;
static constexpr int Cln = 1024;
static constexpr int Hn  = 16;
static constexpr int Dh  = 64;

// fp32 -> bf16 (round to nearest even), raw bits
static __device__ __forceinline__ unsigned short f2bf(float f) {
  unsigned int u = __float_as_uint(f);
  unsigned int r = ((u >> 16) & 1u) + 0x7fffu;
  return (unsigned short)((u + r) >> 16);
}

// Build a 16-element bf16 fragment from two contiguous 8-element chunks.
// Per CDNA5 ISA (CDNA5 ISA, 16-bit A 16x32): lane L holds row M=L%16,
// elements 0..7 = K in [hi*8, hi*8+7], elements 8..15 = K in [16+hi*8 ...],
// hi = L/16.  B mirrors this with N on the lane index.
static __device__ __forceinline__ v16bf load_frag(const unsigned short* p0,
                                                  const unsigned short* p1) {
  v16bf a;
  __builtin_memcpy(&a, p0, 16);
  __builtin_memcpy(reinterpret_cast<unsigned short*>(&a) + 8, p1, 16);
  return a;
}

static __device__ __forceinline__ v8f wmma_bf16(v16bf a, v16bf b, v8f c) {
  // (neg_a, A, neg_b, B, c_mod, C, reuse_a, reuse_b)
  return __builtin_amdgcn_wmma_f32_16x16x32_bf16(false, a, false, b,
                                                 (short)0, c, false, false);
}

// ---------------------------------------------------------------------------
// GEMM: Y = X @ W^T (+bias).  X:[M=8192,K=1024] fp32 or bf16, W:[N=1024,K].
// Block = 256 threads (8 waves), tile 128(M) x 64(N), k-step 32.
// Wave (mw = wave&3, nw = wave>>2) owns a 32x32 patch: 2 A-frags x 2 B-frags
// -> 4 WMMA per k-step with 2x fragment reuse.
// QKV_STORE scatters bf16 output into [B,H,T,D]; else fp32 [M,N] (+bias).
// ---------------------------------------------------------------------------
template <bool IN_F32, bool QKV_STORE, bool ADD_BIAS>
__global__ __launch_bounds__(256) void gemm_xwT(const void* __restrict__ Xv,
                                                const float* __restrict__ W,
                                                const float* __restrict__ bias,
                                                void* __restrict__ Yv) {
  constexpr int LDT = 40;  // 32 + 8 pad: 80B rows -> 16B aligned, no bank dup
  __shared__ unsigned short Xs[128][LDT];
  __shared__ unsigned short Ws[64][LDT];

  const int tid  = threadIdx.x;
  const int wave = tid >> 5;
  const int lane = tid & 31;
  const int lr   = lane & 15;
  const int hi   = lane >> 4;
  const int mw   = wave & 3;
  const int nw   = wave >> 2;
  const int m0   = blockIdx.x * 128;
  const int n0   = blockIdx.y * 64;

  const int xRow = tid >> 1;         // 0..127
  const int xCol = (tid & 1) * 16;   // 0,16
  const int wRow = tid >> 2;         // 0..63
  const int wCol = (tid & 3) * 8;    // 0,8,16,24

  v8f acc00 = {}, acc01 = {}, acc10 = {}, acc11 = {};

  for (int k0 = 0; k0 < Cln; k0 += 32) {
    // ---- cooperative tile loads (convert fp32 -> bf16 on the fly) ----
    if (IN_F32) {
      const float* X = (const float*)Xv;
      const float* src = X + (size_t)(m0 + xRow) * Cln + k0 + xCol;
      unsigned short t16[16];
#pragma unroll
      for (int i = 0; i < 16; ++i) t16[i] = f2bf(src[i]);
      __builtin_memcpy(&Xs[xRow][xCol], t16, 32);
    } else {
      const unsigned short* X = (const unsigned short*)Xv;
      __builtin_memcpy(&Xs[xRow][xCol],
                       X + (size_t)(m0 + xRow) * Cln + k0 + xCol, 32);
    }
    {
      const float* src = W + (size_t)(n0 + wRow) * Cln + k0 + wCol;
      unsigned short t8[8];
#pragma unroll
      for (int i = 0; i < 8; ++i) t8[i] = f2bf(src[i]);
      __builtin_memcpy(&Ws[wRow][wCol], t8, 16);
    }
    __syncthreads();

    // ---- fragments + WMMA (2x2 register tile) ----
    const int ar0 = mw * 32 + lr;
    const int ar1 = mw * 32 + 16 + lr;
    v16bf a0 = load_frag(&Xs[ar0][hi * 8], &Xs[ar0][16 + hi * 8]);
    v16bf a1 = load_frag(&Xs[ar1][hi * 8], &Xs[ar1][16 + hi * 8]);
    const int br0 = nw * 32 + lr;
    const int br1 = nw * 32 + 16 + lr;
    v16bf b0 = load_frag(&Ws[br0][hi * 8], &Ws[br0][16 + hi * 8]);
    v16bf b1 = load_frag(&Ws[br1][hi * 8], &Ws[br1][16 + hi * 8]);
    acc00 = wmma_bf16(a0, b0, acc00);
    acc01 = wmma_bf16(a0, b1, acc01);
    acc10 = wmma_bf16(a1, b0, acc10);
    acc11 = wmma_bf16(a1, b1, acc11);
    __syncthreads();
  }

  // ---- epilogue: C layout = VGPR r -> row (r + 8*hi), col = lane%16 ----
#pragma unroll
  for (int i = 0; i < 2; ++i) {
#pragma unroll
    for (int j = 0; j < 2; ++j) {
#pragma unroll
      for (int r = 0; r < 8; ++r) {
        float v;
        if (i == 0 && j == 0) v = acc00[r];
        else if (i == 0 && j == 1) v = acc01[r];
        else if (i == 1 && j == 0) v = acc10[r];
        else v = acc11[r];
        const int m = m0 + mw * 32 + i * 16 + r + 8 * hi;
        const int n = n0 + nw * 32 + j * 16 + lr;
        if (QKV_STORE) {
          const int bb = m >> 11, t = m & (Tln - 1);
          const int h = n >> 6, d = n & (Dh - 1);
          ((unsigned short*)Yv)[(((size_t)bb * Hn + h) * Tln + t) * Dh + d] =
              f2bf(v);
        } else {
          float o = v;
          if (ADD_BIAS) o += bias[n];
          ((float*)Yv)[(size_t)m * Cln + n] = o;
        }
      }
    }
  }
}

// ---------------------------------------------------------------------------
// Fused flash attention.  Grid: (B*H, T/128). Block = 256 threads (8 waves),
// each wave owns a 16-row query tile; K key blocks of 32 staged into LDS via
// the CDNA5 async-to-LDS path (ASYNCcnt), V transposed through registers so
// PV B-fragments are contiguous.  Online softmax in fp32.
// ---------------------------------------------------------------------------
__global__ __launch_bounds__(256) void attn_kernel(
    const unsigned short* __restrict__ Q, const unsigned short* __restrict__ K,
    const unsigned short* __restrict__ V, const int* __restrict__ mask,
    const int* __restrict__ causal_flag, unsigned short* __restrict__ O) {
  constexpr int LK = 72;  // 64 + 8 pad (144B rows)
  constexpr int LV = 40;  // 32 + 8 pad (80B rows)
  constexpr int LP = 40;
  __shared__ unsigned short Ks[32][LK];   // K block, row-major [k][d]
  __shared__ unsigned short Vs[Dh][LV];   // V block, transposed [d][k]
  __shared__ unsigned short Ps[8][16][LP];  // per-wave P (C-layout -> A-layout)

  const int tid  = threadIdx.x;
  const int wave = tid >> 5;
  const int lane = tid & 31;
  const int lr   = lane & 15;
  const int hi   = lane >> 4;
  const int bh   = blockIdx.x;   // b*H + h
  const int b    = bh >> 4;
  const int qw   = blockIdx.y * 128 + wave * 16;  // wave's query row base
  const bool causal = (causal_flag[0] != 0);

  // Q fragments (rows lr, D split into two k=32 chunks), loaded once.
  const unsigned short* qrow_p = Q + ((size_t)bh * Tln + qw + lr) * Dh;
  v16bf aq0 = load_frag(qrow_p + hi * 8, qrow_p + 16 + hi * 8);
  v16bf aq1 = load_frag(qrow_p + 32 + hi * 8, qrow_p + 48 + hi * 8);

  v8f o0 = {}, o1 = {}, o2 = {}, o3 = {};
  float mrow[8], lrow[8];
#pragma unroll
  for (int r = 0; r < 8; ++r) { mrow[r] = -__builtin_inff(); lrow[r] = 0.f; }

  const int nkb = causal ? (blockIdx.y + 1) * 4 : (Tln / 32);
  const unsigned short* Kb = K + (size_t)bh * Tln * Dh;
  const unsigned short* Vb = V + (size_t)bh * Tln * Dh;

  const int cRow = tid >> 3;        // 0..31
  const int cCol = (tid & 7) * 8;   // 0..56

  for (int kb = 0; kb < nkb; ++kb) {
    const int k0 = kb * 32;
    // ---- K tile: direct memory -> LDS async copy (ASYNCcnt path) ----
    {
      const unsigned lds_k = (unsigned)(size_t)&Ks[cRow][cCol];
      const unsigned long long gk =
          (unsigned long long)(size_t)(Kb + (size_t)(k0 + cRow) * Dh + cCol);
      asm volatile("global_load_async_to_lds_b128 %0, %1, off"
                   :: "v"(lds_k), "v"(gk)
                   : "memory");
    }
    // ---- V tile: load + transpose through registers ----
    {
      unsigned short vt[8];
      __builtin_memcpy(vt, Vb + (size_t)(k0 + cRow) * Dh + cCol, 16);
#pragma unroll
      for (int i = 0; i < 8; ++i) Vs[cCol + i][cRow] = vt[i];
    }
    if (kb + 1 < nkb) {  // prefetch next tiles (global_prefetch_b8)
      __builtin_prefetch(Kb + (size_t)(k0 + 32 + cRow) * Dh + cCol, 0, 0);
      __builtin_prefetch(Vb + (size_t)(k0 + 32 + cRow) * Dh + cCol, 0, 0);
    }
    asm volatile("s_wait_asynccnt 0x0" ::: "memory");
    __syncthreads();

    if (!causal || k0 <= qw + 15) {  // uniform per wave (EXEC stays all-1s)
      // ---- S = Q K^T for two 16-col key subtiles ----
      v16bf bka = load_frag(&Ks[lr][hi * 8], &Ks[lr][16 + hi * 8]);
      v16bf bkb = load_frag(&Ks[lr][32 + hi * 8], &Ks[lr][48 + hi * 8]);
      v8f s0 = {};
      s0 = wmma_bf16(aq0, bka, s0);
      s0 = wmma_bf16(aq1, bkb, s0);
      v16bf bkc = load_frag(&Ks[16 + lr][hi * 8], &Ks[16 + lr][16 + hi * 8]);
      v16bf bkd = load_frag(&Ks[16 + lr][32 + hi * 8], &Ks[16 + lr][48 + hi * 8]);
      v8f s1 = {};
      s1 = wmma_bf16(aq0, bkc, s1);
      s1 = wmma_bf16(aq1, bkd, s1);

      const int kc0 = k0 + lr, kc1 = k0 + 16 + lr;
      const int mk0 = mask[b * Tln + kc0];
      const int mk1 = mask[b * Tln + kc1];
      constexpr float sc = 0.125f;  // 1/sqrt(64)

      // ---- online softmax; rows live in 16-lane halves (C layout) ----
#pragma unroll
      for (int r = 0; r < 8; ++r) {
        const int qrow = qw + r + 8 * hi;
        float x0 = s0[r] * sc, x1 = s1[r] * sc;
        if (mk0 == 0 || (causal && kc0 > qrow)) x0 = -__builtin_inff();
        if (mk1 == 0 || (causal && kc1 > qrow)) x1 = -__builtin_inff();
        float mx = fmaxf(x0, x1);
#pragma unroll
        for (int sh = 1; sh < 16; sh <<= 1)
          mx = fmaxf(mx, __shfl_xor(mx, sh, 32));
        const float mnew = fmaxf(mrow[r], mx);
        const float mz = (mnew > -1e30f) ? mnew : 0.f;
        const float p0 = __expf(x0 - mz);
        const float p1 = __expf(x1 - mz);
        float ls = p0 + p1;
#pragma unroll
        for (int sh = 1; sh < 16; sh <<= 1) ls += __shfl_xor(ls, sh, 32);
        const float alpha = __expf(mrow[r] - mz);
        lrow[r] = lrow[r] * alpha + ls;
        mrow[r] = mnew;
        o0[r] *= alpha; o1[r] *= alpha; o2[r] *= alpha; o3[r] *= alpha;
        const int pr = r + 8 * hi;
        Ps[wave][pr][lr] = f2bf(p0);
        Ps[wave][pr][16 + lr] = f2bf(p1);
      }

      // ---- O += P V (k = 32 key block; 4 D-subtiles). DS in-order per wave.
      v16bf ap = load_frag(&Ps[wave][lr][hi * 8], &Ps[wave][lr][16 + hi * 8]);
      v16bf bv0 = load_frag(&Vs[lr][hi * 8], &Vs[lr][16 + hi * 8]);
      v16bf bv1 = load_frag(&Vs[16 + lr][hi * 8], &Vs[16 + lr][16 + hi * 8]);
      v16bf bv2 = load_frag(&Vs[32 + lr][hi * 8], &Vs[32 + lr][16 + hi * 8]);
      v16bf bv3 = load_frag(&Vs[48 + lr][hi * 8], &Vs[48 + lr][16 + hi * 8]);
      o0 = wmma_bf16(ap, bv0, o0);
      o1 = wmma_bf16(ap, bv1, o1);
      o2 = wmma_bf16(ap, bv2, o2);
      o3 = wmma_bf16(ap, bv3, o3);
    }
    __syncthreads();
  }

  // ---- normalize and scatter to [B, T, H*D] (bf16) for the output GEMM ----
  const int h = bh & (Hn - 1);
#pragma unroll
  for (int r = 0; r < 8; ++r) {
    const float rl = (lrow[r] > 0.f) ? (1.f / lrow[r]) : 0.f;
    const int qrow = qw + r + 8 * hi;
    const size_t base = ((size_t)b * Tln + qrow) * Cln + h * Dh;
    O[base + 0 + lr]  = f2bf(o0[r] * rl);
    O[base + 16 + lr] = f2bf(o1[r] * rl);
    O[base + 32 + lr] = f2bf(o2[r] * rl);
    O[base + 48 + lr] = f2bf(o3[r] * rl);
  }
}

// ---------------------------------------------------------------------------
extern "C" void kernel_launch(void* const* d_in, const int* in_sizes, int n_in,
                              void* d_out, int out_size, void* d_ws,
                              size_t ws_size, hipStream_t stream) {
  (void)in_sizes; (void)n_in; (void)out_size; (void)ws_size;
  const float* query  = (const float*)d_in[0];
  const float* key    = (const float*)d_in[1];
  const float* value  = (const float*)d_in[2];
  const int*   mask   = (const int*)d_in[3];
  const int*   causal = (const int*)d_in[4];
  const float* Wq     = (const float*)d_in[5];
  const float* Wk     = (const float*)d_in[6];
  const float* Wv     = (const float*)d_in[7];
  const float* Wo     = (const float*)d_in[8];
  const float* bo     = (const float*)d_in[9];
  float* out          = (float*)d_out;

  const size_t SZ = (size_t)Bsz * Tln * Cln;  // 8M bf16 elements each
  unsigned short* q_ws = (unsigned short*)d_ws;
  unsigned short* k_ws = q_ws + SZ;
  unsigned short* v_ws = q_ws + 2 * SZ;
  unsigned short* a_ws = q_ws + 3 * SZ;

  dim3 gG(Bsz * Tln / 128, Cln / 64);  // (64, 16)
  gemm_xwT<true, true, false><<<gG, 256, 0, stream>>>(query, Wq, nullptr, q_ws);
  gemm_xwT<true, true, false><<<gG, 256, 0, stream>>>(key,   Wk, nullptr, k_ws);
  gemm_xwT<true, true, false><<<gG, 256, 0, stream>>>(value, Wv, nullptr, v_ws);

  dim3 gA(Bsz * Hn, Tln / 128);  // (64, 16)
  attn_kernel<<<gA, 256, 0, stream>>>(q_ws, k_ws, v_ws, mask, causal, a_ws);

  gemm_xwT<false, false, true><<<gG, 256, 0, stream>>>(a_ws, Wo, bo, out);
}
